// YOLOLayer_78898549228208
// MI455X (gfx1250) — matrix operand order repaired and checked by reading, not compiled, not run on hardware
//
#include <hip/hip_runtime.h>
#include <hip/hip_bf16.h>

// ---------------------------------------------------------------------------
// YOLO layer for MI455X (gfx1250): f16 WMMA GEMM (v_wmma_f32_16x16x32_f16)
// + fused decode, then loss kernels with deterministic fixed-order reductions.
// Problem: B=16, C_IN=256, N=52, A=3, NC=80, O=255, M=400, L=B*A*N*N=129792
// ---------------------------------------------------------------------------

typedef __attribute__((ext_vector_type(16))) _Float16 v16h;
typedef __attribute__((ext_vector_type(8)))  _Float16 v8h;
typedef __attribute__((ext_vector_type(8)))  float    v8f;

#define NNSQ   2704            // N*N
#define ANN    8112            // A*N*N
#define LCELLS 129792          // B*A*N*N
#define PRED_ROWS 129792
#define ASTRIDE 264            // LDS row stride in halfs (16B-aligned, conflict-reduced)

// ALL_ANCHORS / STRIDE
__device__ __constant__ float AAW[9] = {1.25f, 2.0f, 4.125f, 2.875f, 7.75f, 7.375f, 14.5f, 19.5f, 46.625f};
__device__ __constant__ float AAH[9] = {1.625f, 3.75f, 2.875f, 7.625f, 5.625f, 14.875f, 11.25f, 24.75f, 40.75f};

__device__ inline float sigf(float v)  { return 1.0f / (1.0f + __expf(-v)); }
__device__ inline float clogf_(float p){ return fmaxf(__logf(p), -100.0f); }
__device__ inline float bce_(float p, float t) {
  return -(t * clogf_(p) + (1.0f - t) * clogf_(1.0f - p));
}

// --------------------------- weights fp32 -> f16 (pad row 255 with zeros) ---
__global__ __launch_bounds__(256) void conv_w_to_f16(const float* __restrict__ w,
                                                     _Float16* __restrict__ wf16) {
  int e = blockIdx.x * 256 + threadIdx.x;      // 0..65535 (256x256)
  int o = e >> 8;
  wf16[e] = (o < 255) ? (_Float16)w[e] : (_Float16)0.0f;
}

__global__ __launch_bounds__(256) void zero_flags(int* __restrict__ f) {
  f[blockIdx.x * 256 + threadIdx.x] = 0;
}

// --------------------------- GEMM + YOLO decode -----------------------------
// grid 676 blocks of 256 threads; each block: 64 positions x 255 outputs.
__global__ __launch_bounds__(256) void gemm_decode(const float* __restrict__ x,
                                                   const _Float16* __restrict__ wf16,
                                                   const float* __restrict__ conv_b,
                                                   float* __restrict__ pred) {
  __shared__ _Float16 As[64 * ASTRIDE];        // [pos][chan] f16, ~33.8 KB
  const int t    = threadIdx.x;
  const int pos0 = blockIdx.x * 64;

  // Stage A tile: 64 positions x 256 channels, converted to f16.
  for (int it = 0; it < 64; ++it) {
    int e = t + 256 * it;                      // e = c*64 + q  (coalesced in q)
    int q = e & 63, c = e >> 6;
    int p = pos0 + q;
    int b = p / NNSQ, s = p - b * NNSQ;
    As[q * ASTRIDE + c] = (_Float16)x[(b * 256 + c) * NNSQ + s];
  }
  __syncthreads();

  const int wave  = t >> 5, lane = t & 31;
  const int prow  = wave & 3;                  // 16-position sub-tile 0..3
  const int og    = wave >> 2;                 // output half: 0 -> o 0..127, 1 -> 128..255
  const int lhalf = lane >> 4, lmod = lane & 15;

  v8f acc[8];
  const v8f vzero = {0.f, 0.f, 0.f, 0.f, 0.f, 0.f, 0.f, 0.f};
#pragma unroll
  for (int i = 0; i < 8; ++i) acc[i] = vzero;

  const _Float16* Abase = As + (prow * 16 + lmod) * ASTRIDE;
  const int obase = og * 128 + lmod;

#pragma unroll
  for (int k0 = 0; k0 < 256; k0 += 32) {
    // A 16x32 layout (matches B): lane holds K = k0+16*lhalf..+7 (elems 0..7)
    // and K = k0+16*lhalf+8..+15 (elems 8..15).
    v8h lo = *(const v8h*)(Abase + k0 + 16 * lhalf);
    v8h hi = *(const v8h*)(Abase + k0 + 16 * lhalf + 8);
    v16h af = __builtin_shufflevector(lo, hi, 0,1,2,3,4,5,6,7,8,9,10,11,12,13,14,15);
#pragma unroll
    for (int nt = 0; nt < 8; ++nt) {
      // B 32x16: lane = column o, 16 consecutive K starting at k0 + 16*lhalf.
      int o = obase + nt * 16;
      v16h bf = *(const v16h*)(wf16 + o * 256 + k0 + 16 * lhalf);
      acc[nt] = __builtin_amdgcn_wmma_f32_16x16x32_f16(
          false, af, false, bf, (short)0, acc[nt], false, false);
    }
  }

  // C/D layout: vgpr r holds M = 8*lhalf + r, column N = lmod.
  int pb[8], ps[8], pi_[8], pj[8];
#pragma unroll
  for (int r = 0; r < 8; ++r) {
    int p = pos0 + prow * 16 + lhalf * 8 + r;
    int b = p / NNSQ; int s = p - b * NNSQ;
    int i2 = s / 52;
    pb[r] = b; ps[r] = s; pi_[r] = i2; pj[r] = s - i2 * 52;
  }

#pragma unroll
  for (int nt = 0; nt < 8; ++nt) {
    int o = obase + nt * 16;
    if (o >= 255) continue;                    // padding column
    float bias = conv_b[o];
    int a = o / 85, c = o - a * 85;
#pragma unroll
    for (int r = 0; r < 8; ++r) {
      float v = acc[nt][r] + bias;
      float pv;
      if (c == 0)      pv = (sigf(v) + (float)pj[r]) * 8.0f;          // bx*STRIDE
      else if (c == 1) pv = (sigf(v) + (float)pi_[r]) * 8.0f;         // by*STRIDE
      else if (c == 2) pv = __expf(v) * (AAW[a] * 8.0f);              // bw*STRIDE
      else if (c == 3) pv = __expf(v) * (AAH[a] * 8.0f);              // bh*STRIDE
      else             pv = sigf(v);                                  // obj/cls
      pred[(size_t)((pb[r] * 3 + a) * NNSQ + ps[r]) * 85 + c] = pv;
    }
  }
}

// --------------------------- label -> cell scatter (tgt_obj / obj_set) ------
__global__ __launch_bounds__(256) void label_scatter(const float* __restrict__ labels,
                                                     int* __restrict__ obj_flag) {
  int m = blockIdx.x * 256 + threadIdx.x;
  if (m >= 400) return;
  int   lbl_b = (int)labels[m * 6 + 0];
  float lx = labels[m * 6 + 2] * 52.f, ly = labels[m * 6 + 3] * 52.f;
  float lw = labels[m * 6 + 4] * 52.f, lh = labels[m * 6 + 5] * 52.f;
  int bi = 0; float bestv = -1.0f;
#pragma unroll
  for (int k = 0; k < 9; ++k) {               // anchor IoU (co-centered boxes)
    float inter = fminf(lw, AAW[k]) * fminf(lh, AAH[k]);
    float iou   = inter / (lw * lh + AAW[k] * AAH[k] - inter);
    if (iou > bestv) { bestv = iou; bi = k; } // first-max like jnp.argmax
  }
  if (bi < 3) {                               // lbl_mask
    int li = (int)floorf(ly), lj = (int)floorf(lx);
    int idx = ANN * lbl_b + NNSQ * (bi % 3) + 52 * li + lj;
    atomicOr(&obj_flag[idx], 1);
  }
}

// --------------------------- objectness loss over all cells -----------------
__global__ __launch_bounds__(256) void obj_loss(const float* __restrict__ pred,
                                                const float* __restrict__ labels,
                                                const int* __restrict__ obj_flag,
                                                float* __restrict__ cpart) {
  __shared__ float lx0[400], ly0[400], lx1[400], ly1[400], lar[400];
  __shared__ float red[256];
  int t = threadIdx.x;
  for (int m = t; m < 400; m += 256) {
    float lb = labels[m * 6 + 0];
    float cx = labels[m * 6 + 2] * 52.f + lb * 52.f;
    float cy = labels[m * 6 + 3] * 52.f + lb * 52.f;
    float w  = labels[m * 6 + 4] * 52.f;
    float h  = labels[m * 6 + 5] * 52.f;
    lx0[m] = cx - w * 0.5f; ly0[m] = cy - h * 0.5f;
    lx1[m] = cx + w * 0.5f; ly1[m] = cy + h * 0.5f;
    lar[m] = w * h;
  }
  __syncthreads();

  int l = blockIdx.x * 256 + t;               // 507*256 == LCELLS exactly
  const float* pr = pred + (size_t)l * 85;
  float boff = (float)(l / ANN) * 52.f;
  float bx = pr[0] * 0.125f + boff, by = pr[1] * 0.125f + boff;
  float bw = pr[2] * 0.125f,        bh = pr[3] * 0.125f;
  float ax0 = bx - bw * 0.5f, ay0 = by - bh * 0.5f;
  float ax1 = bx + bw * 0.5f, ay1 = by + bh * 0.5f;
  float aarea = bw * bh;

  float best = 0.0f;
  for (int m = 0; m < 400; ++m) {
    float iw = fmaxf(fminf(ax1, lx1[m]) - fmaxf(ax0, lx0[m]), 0.0f);
    float ih = fmaxf(fminf(ay1, ly1[m]) - fmaxf(ay0, ly0[m]), 0.0f);
    float inter = iw * ih;
    best = fmaxf(best, inter / (aarea + lar[m] - inter));
  }
  int   set   = obj_flag[l];
  float tgt   = set ? 1.0f : 0.0f;
  float maskv = ((best < 0.7f) || set) ? 1.0f : 0.0f;

  red[t] = maskv * bce_(pr[4], tgt);
  __syncthreads();
  for (int s = 128; s > 0; s >>= 1) { if (t < s) red[t] += red[t + s]; __syncthreads(); }
  if (t == 0) cpart[blockIdx.x] = red[0];
}

// --------------------------- per-label bbox/cls losses (single block) -------
__global__ __launch_bounds__(512) void label_losses(const float* __restrict__ labels,
                                                    const float* __restrict__ pred,
                                                    float* __restrict__ losses) {
  __shared__ float red[512];
  int t = threadIdx.x;
  float s_xy = 0.f, s_wh = 0.f, s_cls = 0.f;
  if (t < 400) {
    int   lbl_b = (int)labels[t * 6 + 0];
    int   cls   = (int)labels[t * 6 + 1];
    float lx = labels[t * 6 + 2] * 52.f, ly = labels[t * 6 + 3] * 52.f;
    float lw = labels[t * 6 + 4] * 52.f, lh = labels[t * 6 + 5] * 52.f;
    int bi = 0; float bestv = -1.0f;
#pragma unroll
    for (int k = 0; k < 9; ++k) {
      float inter = fminf(lw, AAW[k]) * fminf(lh, AAH[k]);
      float iou   = inter / (lw * lh + AAW[k] * AAH[k] - inter);
      if (iou > bestv) { bestv = iou; bi = k; }
    }
    float m = (bi < 3) ? 1.0f : 0.0f;
    int   al = bi % 3;
    float fi = floorf(ly), fj = floorf(lx);
    int   idx = ANN * lbl_b + NNSQ * al + 52 * (int)fi + (int)fj;
    const float* g = pred + (size_t)idx * 85;
    float aw = AAW[al], ah = AAH[al];
    float tx = lx - fj, ty = ly - fi;
    float tw = __logf(lw / aw + 1e-16f), th = __logf(lh / ah + 1e-16f);
    float scale = 2.0f - (lw / 52.f) * (lh / 52.f);
    // recover outf channels from decoded pred (exact inverse transforms)
    float gx = g[0] * 0.125f - fj;
    float gy = g[1] * 0.125f - fi;
    float gw = __logf(g[2] * 0.125f / aw);
    float gh = __logf(g[3] * 0.125f / ah);
    s_xy = m * scale * (bce_(gx, tx) + bce_(gy, ty));
    s_wh = m * scale * 0.5f * ((gw - tw) * (gw - tw) + (gh - th) * (gh - th));
    float c = 0.f;
    for (int cc = 0; cc < 80; ++cc)
      c += bce_(g[5 + cc], (cc == cls) ? 1.0f : 0.0f);
    s_cls = m * c;
  }
  red[t] = s_xy; __syncthreads();
  for (int s = 256; s > 0; s >>= 1) { if (t < s) red[t] += red[t + s]; __syncthreads(); }
  if (t == 0) losses[0] = red[0]; __syncthreads();
  red[t] = s_wh; __syncthreads();
  for (int s = 256; s > 0; s >>= 1) { if (t < s) red[t] += red[t + s]; __syncthreads(); }
  if (t == 0) losses[1] = red[0]; __syncthreads();
  red[t] = s_cls; __syncthreads();
  for (int s = 256; s > 0; s >>= 1) { if (t < s) red[t] += red[t + s]; __syncthreads(); }
  if (t == 0) losses[3] = red[0];
}

// --------------------------- deterministic final obj-loss reduction ---------
__global__ __launch_bounds__(512) void reduce_obj(const float* __restrict__ cpart,
                                                  float* __restrict__ losses) {
  __shared__ float red[512];
  int t = threadIdx.x;
  red[t] = (t < 507) ? cpart[t] : 0.0f;
  __syncthreads();
  for (int s = 256; s > 0; s >>= 1) { if (t < s) red[t] += red[t + s]; __syncthreads(); }
  if (t == 0) losses[2] = red[0];
}

// ---------------------------------------------------------------------------
extern "C" void kernel_launch(void* const* d_in, const int* in_sizes, int n_in,
                              void* d_out, int out_size, void* d_ws, size_t ws_size,
                              hipStream_t stream) {
  const float* x      = (const float*)d_in[0];
  const float* conv_w = (const float*)d_in[1];
  const float* conv_b = (const float*)d_in[2];
  const float* labels = (const float*)d_in[3];

  float* pred   = (float*)d_out;
  float* losses = (float*)d_out + (out_size - 4);   // pred is 129792*85 floats

  // workspace layout
  _Float16* wf16     = (_Float16*)d_ws;                               // 131072 B
  int*      obj_flag = (int*)((char*)d_ws + 131072);                  // 519168 B
  float*    cpart    = (float*)((char*)d_ws + 131072 + 519168);       // 507 floats

  conv_w_to_f16<<<256, 256, 0, stream>>>(conv_w, wf16);
  zero_flags  <<<507, 256, 0, stream>>>(obj_flag);
  gemm_decode <<<676, 256, 0, stream>>>(x, wf16, conv_b, pred);
  label_scatter<<<2,  256, 0, stream>>>(labels, obj_flag);
  obj_loss    <<<507, 256, 0, stream>>>(pred, labels, obj_flag, cpart);
  label_losses<<<1,   512, 0, stream>>>(labels, pred, losses);
  reduce_obj  <<<1,   512, 0, stream>>>(cpart, losses);
}
